// SelfAxialAttention_57363583206031
// MI455X (gfx1250) — matrix operand order repaired and checked by reading, not compiled, not run on hardware
//
#include <hip/hip_runtime.h>

typedef __attribute__((ext_vector_type(16))) __bf16 v16bf;
typedef __attribute__((ext_vector_type(8)))  __bf16 v8bf;
typedef __attribute__((ext_vector_type(8)))  float  v8f;

// Problem constants: B=16, C=256, H=96, W=96
#define NB   16
#define NC   256
#define NH   96
#define NW   96
#define OUT_ELEMS (16*256*96*96)      // 37748736
#define SLICE_OUT (256*96*96)         // per-batch out stride 2359296

// ---------------- WMMA fragment helpers (ISA 7.12.2 layouts) ----------------

// A-matrix 16x32 bf16 fragment from row-major S[m][k], leading dim ld.
// lane<16: M=lane, K = {0..7, 16..23}; lane>=16: same M, K += 8.
__device__ __forceinline__ v16bf frag_a(const __bf16* S, int ld, int m0, int k0, int lane) {
    const int r = lane & 15, hf = lane >> 4;
    const __bf16* row = S + (size_t)(m0 + r) * ld + k0 + 8 * hf;
    v8bf lo = *(const v8bf*)(row);        // K = 8hf + 0..7
    v8bf hi = *(const v8bf*)(row + 16);   // K = 8hf + 16..23
    return __builtin_shufflevector(lo, hi, 0,1,2,3,4,5,6,7,8,9,10,11,12,13,14,15);
}

// B-matrix 32x16 bf16 fragment, with B[k][n] supplied as S[n][k] (row n, ld).
// lane<16: N=lane, K = {0..7, 16..23}; lane>=16: N=lane-16, K += 8 (pairs packed per VGPR).
__device__ __forceinline__ v16bf frag_b(const __bf16* S, int ld, int k0, int n0, int lane) {
    const int r = lane & 15, hf = lane >> 4;
    const __bf16* row = S + (size_t)(n0 + r) * ld + k0 + 8 * hf;
    v8bf lo = *(const v8bf*)(row);
    v8bf hi = *(const v8bf*)(row + 16);
    return __builtin_shufflevector(lo, hi, 0,1,2,3,4,5,6,7,8,9,10,11,12,13,14,15);
}

__device__ __forceinline__ v8f wmma_bf16(v16bf a, v16bf b, v8f c) {
    return __builtin_amdgcn_wmma_f32_16x16x32_bf16(false, a, false, b, (short)0, c, false, false);
}

// ---------------- Prep: weights -> bf16, positional encoding -> f32 ----------------

__global__ void prep_kernel(const float* __restrict__ wq, const float* __restrict__ wk,
                            const float* __restrict__ wv, __bf16* __restrict__ wb,
                            float* __restrict__ pe) {
    int i = blockIdx.x * 256 + threadIdx.x;
    if (i < 3 * 65536) {
        float v = (i < 65536) ? wq[i] : (i < 131072) ? wk[i - 65536] : wv[i - 131072];
        wb[i] = (__bf16)v;
    } else if (i < 3 * 65536 + NC * NH) {
        int p = i - 3 * 65536;
        int c = p / NH, h = p % NH;
        float val;
        if (c < 128) {
            val = (c & 1) ? 1.0f : 0.0f;              // sin(0)=0 / cos(0)=1 quirk
        } else {
            int t = (c - 128) >> 1;
            float dv  = expf(-(float)(2 * t) * 0.07195578430564880f); // ln(1e4)/128
            float arg = (float)h * dv;
            val = (c & 1) ? cosf(arg) : sinf(arg);
        }
        pe[c * NH + h] = val;
    }
}

// ---------------- Fused per-slice axial attention ----------------

struct SmemA {               // phases 1-2
    __bf16 Vs [NC * NH];     // [c][j]
    __bf16 QT [NH * NC];     // [h][o]
    __bf16 KT [NH * NC];     // [h][o]
    __bf16 XsT[NH * NC];     // [h][c]
};
struct SmemB {               // phases 3-5 (E/AttB alias dead XsT)
    __bf16 Vs [NC * NH];
    __bf16 QT [NH * NC];
    __bf16 KT [NH * NC];
    float  E  [NH * NH];
    __bf16 AttB[NH * NH];
};
union SmemU { SmemA a; SmemB b; };

__global__ __launch_bounds__(256)
void axial_attn_kernel(const float* __restrict__ x,
                       const __bf16* __restrict__ wq, const __bf16* __restrict__ wk,
                       const __bf16* __restrict__ wv, const float* __restrict__ pe,
                       float* __restrict__ out, float* __restrict__ att) {
    __shared__ __align__(16) SmemU sm;
    __bf16* Vs   = sm.a.Vs;
    __bf16* QT   = sm.a.QT;
    __bf16* KT   = sm.a.KT;
    __bf16* XsT  = sm.a.XsT;
    float*  Esm  = sm.b.E;
    __bf16* AttB = sm.b.AttB;

    const int tid  = threadIdx.x;
    const int wave = tid >> 5, lane = tid & 31;
    const int n16  = lane & 15, half = lane >> 4;
    const int w    = blockIdx.x;     // 0..95
    const int bb   = blockIdx.y;     // 0..15
    const int slice = bb * NW + w;

    const v8f vzero = {0.f,0.f,0.f,0.f,0.f,0.f,0.f,0.f};

    // ---- Phase 1: XsT[h][c] = bf16(x[bb,c,h,w] + pe[c,h]) ----
    for (int i = tid; i < NC * NH; i += 256) {
        int h = i >> 8, c = i & 255;
        float xv = x[(((size_t)bb * NC + c) * NH + h) * NW + w];
        XsT[h * NC + c] = (__bf16)(xv + pe[c * NH + h]);
    }
    __syncthreads();

    // ---- Phase 2: Q,K,V = W{q,k,v} * Xs  (16x16x32 WMMA, K-loop = 256/32) ----
    for (int sel = 0; sel < 3; ++sel) {
        const __bf16* Wt = (sel == 0) ? wq : (sel == 1) ? wk : wv;
        for (int ot = wave; ot < 16; ot += 8) {       // o-tile rows, 2 per wave
            v8f acc[6];
            #pragma unroll
            for (int ht = 0; ht < 6; ++ht) acc[ht] = vzero;
            for (int kc = 0; kc < 8; ++kc) {
                v16bf a = frag_a(Wt, NC, ot * 16, kc * 32, lane);
                #pragma unroll
                for (int ht = 0; ht < 6; ++ht) {
                    v16bf bv = frag_b(XsT, NC, kc * 32, ht * 16, lane);
                    acc[ht] = wmma_bf16(a, bv, acc[ht]);
                }
            }
            if (sel < 2) {  // store transposed [h][o] -> one 16B ds_store per lane
                __bf16* Dst = (sel == 0) ? QT : KT;
                #pragma unroll
                for (int ht = 0; ht < 6; ++ht) {
                    v8bf t;
                    #pragma unroll
                    for (int v = 0; v < 8; ++v) t[v] = (__bf16)acc[ht][v];
                    *(v8bf*)&Dst[(ht * 16 + n16) * NC + ot * 16 + 8 * half] = t;
                }
            } else {        // V stored row-major [c][j]
                #pragma unroll
                for (int ht = 0; ht < 6; ++ht)
                    #pragma unroll
                    for (int v = 0; v < 8; ++v)
                        Vs[(ot * 16 + v + 8 * half) * NH + ht * 16 + n16] = (__bf16)acc[ht][v];
            }
        }
    }
    __syncthreads();

    // ---- Phase 3: E = Q^T K  (36 tiles of 16x16, K-loop = 256/32) ----
    for (int t = wave; t < 36; t += 8) {
        int it = t / 6, jt = t % 6;
        v8f acc = vzero;
        for (int kc = 0; kc < 8; ++kc) {
            v16bf a  = frag_a(QT, NC, it * 16, kc * 32, lane);   // A[i][o]
            v16bf bv = frag_b(KT, NC, kc * 32, jt * 16, lane);   // B[o][j] via KT[j][o]
            acc = wmma_bf16(a, bv, acc);
        }
        #pragma unroll
        for (int v = 0; v < 8; ++v)
            Esm[(it * 16 + v + 8 * half) * NH + jt * 16 + n16] = acc[v];
    }
    __syncthreads();

    // ---- Phase 4: row softmax (f32), also bf16 copy for the V GEMM ----
    if (tid < NH) {
        float* er = Esm + tid * NH;
        float m = -3.402823466e38f;
        for (int j = 0; j < NH; ++j) m = fmaxf(m, er[j]);
        float s = 0.f;
        for (int j = 0; j < NH; ++j) { float e = __expf(er[j] - m); er[j] = e; s += e; }
        float inv = 1.0f / s;
        for (int j = 0; j < NH; ++j) {
            float a = er[j] * inv;
            er[j] = a;
            AttB[tid * NH + j] = (__bf16)a;
        }
    }
    __syncthreads();

    // attention output (coalesced from LDS)
    float* attp = att + (size_t)slice * (NH * NH);
    for (int i = tid; i < NH * NH; i += 256) attp[i] = Esm[i];

    // ---- Phase 5: out = V * Att^T  (96 tiles, K-loop = 96/32) ----
    for (int t = wave; t < 96; t += 8) {
        int ct = t / 6, it = t % 6;
        v8f acc = vzero;
        for (int kc = 0; kc < 3; ++kc) {
            v16bf a  = frag_a(Vs,  NH, ct * 16, kc * 32, lane);   // V[c][j]
            v16bf bv = frag_b(AttB, NH, kc * 32, it * 16, lane);  // B[j][i] via Att[i][j]
            acc = wmma_bf16(a, bv, acc);
        }
        #pragma unroll
        for (int v = 0; v < 8; ++v) {
            int c = ct * 16 + v + 8 * half;
            int h = it * 16 + n16;
            out[(size_t)bb * SLICE_OUT + (size_t)c * (NH * NW) + h * NW + w] = acc[v];
        }
    }
}

// ---------------- Launch ----------------

extern "C" void kernel_launch(void* const* d_in, const int* in_sizes, int n_in,
                              void* d_out, int out_size, void* d_ws, size_t ws_size,
                              hipStream_t stream) {
    const float* x  = (const float*)d_in[0];
    const float* Wq = (const float*)d_in[1];
    const float* Wk = (const float*)d_in[2];
    const float* Wv = (const float*)d_in[3];

    __bf16* wb = (__bf16*)d_ws;                       // 3 * 65536 bf16
    float*  pe = (float*)((char*)d_ws + 3 * 65536 * 2);  // 256*96 f32

    float* out = (float*)d_out;
    float* att = out + OUT_ELEMS;

    // prep: 3*65536 weight elems + 256*96 pe elems = 221184 -> 864 blocks
    prep_kernel<<<864, 256, 0, stream>>>(Wq, Wk, Wv, wb, pe);

    dim3 grid(NW, NB);   // one workgroup per (w, b) slice; w-adjacent share L2 lines
    axial_attn_kernel<<<grid, 256, 0, stream>>>(x, wb, wb + 65536, wb + 131072,
                                                pe, out, att);
}
